// LinearAttention_57578331570194
// MI455X (gfx1250) — matrix operand order repaired and checked by reading, not compiled, not run on hardware
//
#include <hip/hip_runtime.h>

#define B_  2
#define H_  4
#define T_  2048
#define NH_ 8192
#define D_  256

typedef __attribute__((ext_vector_type(16))) __bf16 v16bf;
typedef __attribute__((ext_vector_type(8)))  float  v8f;

union Frag32B { uint4 u[2]; v16bf v; };

// float -> bf16 bits, round-to-nearest-even
__device__ __forceinline__ unsigned short f2bf(float f) {
  union { float f; unsigned int u; } x;
  x.f = f;
  unsigned int r = x.u + 0x7FFFu + ((x.u >> 16) & 1u);
  return (unsigned short)(r >> 16);
}

// ---------------------------------------------------------------------------
// Pass 1: RoPE + fp32 -> bf16 cast. One thread per (even,odd) pair.
// freq(n) = theta^(-floor(n/2)*2/NH) / (2*pi), theta = 2^16
//        => exp2(-16*n_even/NH) * (1/2pi)
// ---------------------------------------------------------------------------
__global__ void rope_cast_kernel(const float* __restrict__ X,
                                 unsigned short* __restrict__ Y,
                                 long long npairs) {
  long long p = (long long)blockIdx.x * blockDim.x + threadIdx.x;
  if (p >= npairs) return;
  const int nh2 = NH_ / 2;
  int n = (int)(p % nh2) * 2;                 // even pair index
  int t = (int)((p / nh2) % T_);
  float v0 = X[2 * p];
  float v1 = X[2 * p + 1];
  float freq = __builtin_exp2f(-16.0f * (float)n / (float)NH_) * 0.15915494309189535f;
  float ph = (float)t * freq;
  ph = (ph - floorf(ph)) * 6.283185307179586f; // (phases % 1) * 2*pi
  float sn = __sinf(ph);
  float cs = __cosf(ph);
  Y[2 * p]     = f2bf(v0 * cs - v1 * sn);
  Y[2 * p + 1] = f2bf(v1 * cs + v0 * sn);
}

// ---------------------------------------------------------------------------
// Pass 2: V [B,1,T,D] fp32 -> Vt [B,D,T] bf16 (transposed so WMMA B-fragments
// of the second GEMM are per-lane contiguous along the key axis).
// ---------------------------------------------------------------------------
__global__ void vcast_kernel(const float* __restrict__ V,
                             unsigned short* __restrict__ Vt) {
  int idx = blockIdx.x * blockDim.x + threadIdx.x;   // [0, B*T*D)
  int d = idx % D_;
  int t = (idx / D_) % T_;
  int b = idx / (D_ * T_);
  Vt[((size_t)b * D_ + d) * T_ + t] = f2bf(V[idx]);
}

// ---------------------------------------------------------------------------
// Pass 3: streaming causal (strict) attention without softmax.
// Block: 128 threads = 4 waves. Wave w owns query rows
//   m = blockIdx.x*64 + w*16 .. +15 of head bh = blockIdx.y.
// For each 32-key block s_base (only strictly-below-diagonal blocks):
//   S[16x32] = sum_c Q[16x32c] * K^T  via 2 x (NH/32) v_wmma_f32_16x16x32_bf16
//   mask s<t, cvt to bf16, relayout C/D->A via wave-private LDS,
//   O[16x256] += S[16x32] @ Vt      via 16 more bf16 WMMAs.
// ---------------------------------------------------------------------------
__global__ __launch_bounds__(128, 1)
void attn_kernel(const unsigned short* __restrict__ Qbf,
                 const unsigned short* __restrict__ Kbf,
                 const unsigned short* __restrict__ Vt,
                 float* __restrict__ out) {
  __shared__ __align__(16) unsigned short sS[4][16 * 32];

  const int lane = threadIdx.x & 31;
  const int wave = threadIdx.x >> 5;
  const int bh   = blockIdx.y;          // b*H + h
  const int bb   = bh / H_;
  const int m_base = blockIdx.x * 64 + wave * 16;

  const int lhi = lane >> 4;            // 0 or 1 (half-wave)
  const int llo = lane & 15;

  // A-fragment (Q) per-lane base: row M = m_base+llo, K runs contiguous at lhi*16
  const unsigned short* qp = Qbf + ((size_t)bh * T_ + m_base + llo) * NH_ + lhi * 16;
  unsigned short* ws = sS[wave];

  const v8f vzero = {0.f, 0.f, 0.f, 0.f, 0.f, 0.f, 0.f, 0.f};
  v8f oacc[16];
#pragma unroll
  for (int i = 0; i < 16; ++i) oacc[i] = vzero;

  // keys needed: s <= m_base+14  (strict mask, max t in tile = m_base+15)
  for (int s_base = 0; s_base < m_base + 15; s_base += 32) {
    // B-fragment (K^T): column N = key llo, K runs contiguous at lhi*16
    const unsigned short* kp0 = Kbf + ((size_t)bh * T_ + s_base + llo) * NH_ + lhi * 16;
    const unsigned short* kp1 = kp0 + (size_t)16 * NH_;

    v8f sacc0 = vzero, sacc1 = vzero;
    for (int c = 0; c < NH_; c += 32) {
      Frag32B a, b0, b1;
      a.u[0]  = *(const uint4*)(qp + c);
      a.u[1]  = *(const uint4*)(qp + c + 8);
      b0.u[0] = *(const uint4*)(kp0 + c);
      b0.u[1] = *(const uint4*)(kp0 + c + 8);
      b1.u[0] = *(const uint4*)(kp1 + c);
      b1.u[1] = *(const uint4*)(kp1 + c + 8);
      sacc0 = __builtin_amdgcn_wmma_f32_16x16x32_bf16(
          false, a.v, false, b0.v, (short)0, sacc0, false, false);
      sacc1 = __builtin_amdgcn_wmma_f32_16x16x32_bf16(
          false, a.v, false, b1.v, (short)0, sacc1, false, false);
    }

    // mask (keep strictly-lower: s < t), cvt f32->bf16, stage S row-major in LDS
    // C/D layout: lane holds N=llo (key), M = lhi*8 + r
#pragma unroll
    for (int r = 0; r < 8; ++r) {
      int t  = m_base + lhi * 8 + r;
      int s0 = s_base + llo;
      int s1 = s_base + 16 + llo;
      ws[(lhi * 8 + r) * 32 + llo]      = (s0 < t) ? f2bf(sacc0[r]) : (unsigned short)0;
      ws[(lhi * 8 + r) * 32 + 16 + llo] = (s1 < t) ? f2bf(sacc1[r]) : (unsigned short)0;
    }
    asm volatile("s_wait_dscnt 0" ::: "memory");  // wave-local LDS RAW fence

    // reload S in A-fragment layout (M = llo, K runs contiguous at lhi*16)
    Frag32B a2;
    a2.u[0] = *(const uint4*)&ws[llo * 32 + lhi * 16];
    a2.u[1] = *(const uint4*)&ws[llo * 32 + lhi * 16 + 8];

    // O[16 x 256] += S[16 x 32] @ V[32 x 256]
    const unsigned short* vp = Vt + ((size_t)bb * D_ + llo) * T_ + s_base + lhi * 16;
#pragma unroll
    for (int nt = 0; nt < 16; ++nt) {
      Frag32B bv;
      const unsigned short* vpn = vp + (size_t)(nt * 16) * T_;
      bv.u[0] = *(const uint4*)(vpn);
      bv.u[1] = *(const uint4*)(vpn + 8);
      oacc[nt] = __builtin_amdgcn_wmma_f32_16x16x32_bf16(
          false, a2.v, false, bv.v, (short)0, oacc[nt], false, false);
    }
  }

  // write O [B,H,T,D] fp32; C/D layout -> lane llo supplies column d = nt*16+llo
#pragma unroll
  for (int nt = 0; nt < 16; ++nt) {
#pragma unroll
    for (int r = 0; r < 8; ++r) {
      int m = m_base + lhi * 8 + r;
      int d = nt * 16 + llo;
      out[((size_t)bh * T_ + m) * D_ + d] = oacc[nt][r];
    }
  }
}

// ---------------------------------------------------------------------------
extern "C" void kernel_launch(void* const* d_in, const int* in_sizes, int n_in,
                              void* d_out, int out_size, void* d_ws, size_t ws_size,
                              hipStream_t stream) {
  (void)in_sizes; (void)n_in; (void)out_size; (void)ws_size;
  const float* Q = (const float*)d_in[0];
  const float* K = (const float*)d_in[1];
  const float* V = (const float*)d_in[2];
  float* out = (float*)d_out;

  const size_t qk_elems = (size_t)B_ * H_ * T_ * NH_;   // 134,217,728
  unsigned short* Qbf = (unsigned short*)d_ws;
  unsigned short* Kbf = Qbf + qk_elems;
  unsigned short* Vt  = Kbf + qk_elems;                 // B*D*T bf16 (2 MB)

  const long long npairs = (long long)qk_elems / 2;
  const int rope_blocks = (int)((npairs + 255) / 256);
  rope_cast_kernel<<<rope_blocks, 256, 0, stream>>>(Q, Qbf, npairs);
  rope_cast_kernel<<<rope_blocks, 256, 0, stream>>>(K, Kbf, npairs);

  const int v_elems = B_ * T_ * D_;                     // 1,048,576
  vcast_kernel<<<v_elems / 256, 256, 0, stream>>>(V, Vt);

  dim3 grid(T_ / 64, B_ * H_);   // 32 x 8 blocks, 4 query-tiles (waves) each
  attn_kernel<<<grid, 128, 0, stream>>>(Qbf, Kbf, Vt, out);
}
